// FusedSelfAttention_19705309954491
// MI455X (gfx1250) — matrix operand order repaired and checked by reading, not compiled, not run on hardware
//
#include <hip/hip_runtime.h>
#include <hip/hip_bf16.h>

// ---------------------------------------------------------------------------
// Fused self-attention (S=4096, DIM=2048, H=16, Dh=128), f32 in/out,
// all matmuls via v_wmma_f32_16x16x32_f16 (wave32, CDNA5).
// Round 3: 2x-unrolled ping-pong GEMM buffers (no register-copy rotation),
// K-stream prefetch in the attention loop.
// ---------------------------------------------------------------------------

typedef _Float16 f16;
typedef __attribute__((ext_vector_type(16))) _Float16 v16h;
typedef __attribute__((ext_vector_type(8)))  _Float16 v8h;
typedef __attribute__((ext_vector_type(8)))  float    v8f;

#define SEQ      4096
#define DIM      2048
#define NHEADS   16
#define HDIM     128
#define QKV_N    (3 * DIM)          // 6144
#define INV_SQRT_HD 0.08838834764831845f

__device__ __forceinline__ v8f wmma16(v16h a, v16h b, v8f c) {
  // D = A(16x32 f16) * B(32x16 f16) + C(16x16 f32)
  return __builtin_amdgcn_wmma_f32_16x16x32_f16(false, a, false, b,
                                                (short)0, c, false, false);
}

// A fragment: 16x32 tile, row-major source with leading dim `lda` (elements).
// Lane L: m = L&15, half = L>>4. elems 0..7 -> K = 8*half+0..7,
// elems 8..15 -> K = 16+8*half+0..7  (ISA 05_wmma "16-bit A-Matrix 16x32").
__device__ __forceinline__ v16h load_a_frag(const f16* __restrict__ A,
                                            int lda, int lane) {
  const int m = lane & 15, half = lane >> 4;
  const f16* p = A + (size_t)m * lda + 8 * half;
  v8h lo = *(const v8h*)(p);
  v8h hi = *(const v8h*)(p + 16);
  v16h a;
#pragma unroll
  for (int i = 0; i < 8; ++i) { a[i] = lo[i]; a[i + 8] = hi[i]; }
  return a;
}

// B fragment: 32x16 tile from B^T storage [N][K] with leading dim `ldb`.
// Lane L: col n = L&15, elems 0..7 -> K = 8*half+0..7, elems 8..15 -> K = 16+8*half+0..7.
__device__ __forceinline__ v16h load_b_frag(const f16* __restrict__ Bt,
                                            int ldb, int lane) {
  const int n = lane & 15, half = lane >> 4;
  const f16* p = Bt + (size_t)n * ldb + 8 * half;
  v8h lo = *(const v8h*)(p);
  v8h hi = *(const v8h*)(p + 16);
  v16h b;
#pragma unroll
  for (int i = 0; i < 8; ++i) { b[i] = lo[i]; b[i + 8] = hi[i]; }
  return b;
}

// ---------------------------------------------------------------------------
// 1) f32 -> f16 elementwise convert (for x)
// ---------------------------------------------------------------------------
__global__ void cvt_f32_f16(const float* __restrict__ in, f16* __restrict__ out,
                            int n) {
  int i = blockIdx.x * blockDim.x + threadIdx.x;
  if (i < n) out[i] = (f16)in[i];
}

// ---------------------------------------------------------------------------
// 2) f32 [R][C] -> f16 [C][R] tiled transpose (weights into B^T layout)
// ---------------------------------------------------------------------------
__global__ __launch_bounds__(256)
void transpose_f32_f16(const float* __restrict__ in, f16* __restrict__ out,
                       int R, int C) {
  __shared__ float tile[32][33];
  const int c0 = blockIdx.x * 32, r0 = blockIdx.y * 32;
  const int tx = threadIdx.x, ty = threadIdx.y;   // (32, 8)
#pragma unroll
  for (int i = 0; i < 32; i += 8)
    tile[ty + i][tx] = in[(size_t)(r0 + ty + i) * C + c0 + tx];
  __syncthreads();
#pragma unroll
  for (int i = 0; i < 32; i += 8)
    out[(size_t)(c0 + ty + i) * R + r0 + tx] = (f16)tile[tx][ty + i];
}

// ---------------------------------------------------------------------------
// Core 16x64 GEMM strip for one wave, K unrolled x2 with ping-pong buffers:
// loads refill the buffer that was just consumed, so no register rotation
// copies are needed and every load has a full loop body of latency cover.
// ---------------------------------------------------------------------------
__device__ __forceinline__ void gemm_strip(const f16* __restrict__ Abase,
                                           const f16* __restrict__ Btbase,
                                           int lane, v8f acc[4]) {
  v16h a0 = load_a_frag(Abase, DIM, lane);
  v16h b0[4];
#pragma unroll
  for (int t = 0; t < 4; ++t)
    b0[t] = load_b_frag(Btbase + (size_t)t * 16 * DIM, DIM, lane);
  v16h a1 = load_a_frag(Abase + 32, DIM, lane);
  v16h b1[4];
#pragma unroll
  for (int t = 0; t < 4; ++t)
    b1[t] = load_b_frag(Btbase + (size_t)t * 16 * DIM + 32, DIM, lane);

  for (int k0 = 64; k0 < DIM; k0 += 64) {
    __builtin_prefetch(Abase + k0 + 64, 0, 1);   // global_prefetch_b8
    // consume buffer 0 (data from k0-64), refill with k0
#pragma unroll
    for (int t = 0; t < 4; ++t) acc[t] = wmma16(a0, b0[t], acc[t]);
    a0 = load_a_frag(Abase + k0, DIM, lane);
#pragma unroll
    for (int t = 0; t < 4; ++t)
      b0[t] = load_b_frag(Btbase + (size_t)t * 16 * DIM + k0, DIM, lane);
    // consume buffer 1 (data from k0-32), refill with k0+32
#pragma unroll
    for (int t = 0; t < 4; ++t) acc[t] = wmma16(a1, b1[t], acc[t]);
    a1 = load_a_frag(Abase + k0 + 32, DIM, lane);
#pragma unroll
    for (int t = 0; t < 4; ++t)
      b1[t] = load_b_frag(Btbase + (size_t)t * 16 * DIM + k0 + 32, DIM, lane);
  }
  // epilogue: last two k-steps already resident in the buffers
#pragma unroll
  for (int t = 0; t < 4; ++t) acc[t] = wmma16(a0, b0[t], acc[t]);
#pragma unroll
  for (int t = 0; t < 4; ++t) acc[t] = wmma16(a1, b1[t], acc[t]);
}

// ---------------------------------------------------------------------------
// 3) QKV projection: [4096,2048] @ [2048,6144] with scatter into
//    Q[h][s][d] (pre-scaled by 1/sqrt(Dh)), K[h][s][d], V^T[h][d][s].
//    Block = 4 waves, 64x64 tile; wave w owns rows m0..m0+15.
// ---------------------------------------------------------------------------
__global__ __launch_bounds__(128)
void gemm_qkv_kernel(const f16* __restrict__ X, const f16* __restrict__ Wt,
                     f16* __restrict__ Qb, f16* __restrict__ Kb,
                     f16* __restrict__ Vt) {
  const int lane = threadIdx.x & 31;
  const int wave = threadIdx.x >> 5;
  const int n0 = blockIdx.x * 64;
  const int m0 = blockIdx.y * 64 + wave * 16;

  v8f acc[4] = {};
  gemm_strip(X + (size_t)m0 * DIM, Wt + (size_t)n0 * DIM, lane, acc);

  const int cn = lane & 15, half = lane >> 4;
#pragma unroll
  for (int t = 0; t < 4; ++t) {
    const int col = n0 + t * 16 + cn;
    const int which = col >> 11;         // 0=Q 1=K 2=V
    const int r = col & (DIM - 1);
    const int h = r >> 7, d = r & (HDIM - 1);
#pragma unroll
    for (int rr = 0; rr < 8; ++rr) {
      const int s = m0 + rr + 8 * half;
      const float v = acc[t][rr];
      if (which == 0)
        Qb[((size_t)h * SEQ + s) * HDIM + d] = (f16)(v * INV_SQRT_HD);
      else if (which == 1)
        Kb[((size_t)h * SEQ + s) * HDIM + d] = (f16)v;
      else
        Vt[((size_t)h * HDIM + d) * SEQ + s] = (f16)v;
    }
  }
}

// ---------------------------------------------------------------------------
// 4) Flash attention: one wave per (16-query tile, head). Online softmax.
//    Scores: 8 WMMA / 32 keys. V fragments issued right after the score
//    WMMAs (fly during softmax VALU work); next key-chunk's K rows are
//    prefetched lane-spread at the same time. PV: P relaid via LDS
//    (C-frag -> A-frag), 8 WMMA / 32 keys into O accumulators (16 x 128).
// ---------------------------------------------------------------------------
__global__ __launch_bounds__(32)
void attn_kernel(const f16* __restrict__ Qb, const f16* __restrict__ Kb,
                 const f16* __restrict__ Vt, f16* __restrict__ AO) {
  __shared__ f16 plds[16 * 32];
  const int lane = threadIdx.x;
  const int q0 = blockIdx.x * 16;
  const int h  = blockIdx.y;
  const f16* Qh = Qb + (size_t)h * SEQ * HDIM;
  const f16* Kh = Kb + (size_t)h * SEQ * HDIM;
  const f16* Vh = Vt + (size_t)h * HDIM * SEQ;
  const int cn = lane & 15, half = lane >> 4;

  v16h qf[4];
#pragma unroll
  for (int d = 0; d < 4; ++d)
    qf[d] = load_a_frag(Qh + (size_t)q0 * HDIM + d * 32, HDIM, lane);

  v8f o[8] = {};
  float mi[8], li[8];
#pragma unroll
  for (int r = 0; r < 8; ++r) { mi[r] = -1e30f; li[r] = 0.0f; }

  for (int kc = 0; kc < SEQ; kc += 32) {
    // ---- scores: S[16x32] = Q * K^T for keys kc..kc+31
    v8f s0 = {}, s1 = {};
#pragma unroll
    for (int d = 0; d < 4; ++d) {
      v16h kf0 = load_b_frag(Kh + (size_t)kc * HDIM + d * 32, HDIM, lane);
      v16h kf1 = load_b_frag(Kh + (size_t)(kc + 16) * HDIM + d * 32, HDIM, lane);
      s0 = wmma16(qf[d], kf0, s0);
      s1 = wmma16(qf[d], kf1, s1);
    }

    // ---- issue V-fragment loads now: independent of softmax, they fly
    //      while the VALU-heavy exp/reduce section runs.
    v16h vf[8];
#pragma unroll
    for (int t = 0; t < 8; ++t)
      vf[t] = load_b_frag(Vh + (size_t)(t * 16) * SEQ + kc, SEQ, lane);

    // ---- prefetch next key chunk's K rows (lane-spread: one row per lane,
    //      both 128B halves of the 256B row) -> warm for next iteration.
    const f16* knext = Kh + (size_t)(kc + 32 + lane) * HDIM;
    __builtin_prefetch(knext, 0, 1);
    __builtin_prefetch(knext + 64, 0, 1);

    // ---- online softmax: row r (+8*half) lives at VGPR r, cols across lanes
    float alpha[8];
#pragma unroll
    for (int r = 0; r < 8; ++r) {
      float v = fmaxf(s0[r], s1[r]);
#pragma unroll
      for (int m = 1; m < 16; m <<= 1) v = fmaxf(v, __shfl_xor(v, m, 32));
      const float nm = fmaxf(mi[r], v);
      alpha[r] = __expf(mi[r] - nm);
      mi[r] = nm;
      const float p0 = __expf(s0[r] - nm);
      const float p1 = __expf(s1[r] - nm);
      s0[r] = p0; s1[r] = p1;
      float sum = p0 + p1;
#pragma unroll
      for (int m = 1; m < 16; m <<= 1) sum += __shfl_xor(sum, m, 32);
      li[r] = alpha[r] * li[r] + sum;
    }
#pragma unroll
    for (int t = 0; t < 8; ++t)
#pragma unroll
      for (int r = 0; r < 8; ++r) o[t][r] *= alpha[r];

    // ---- relayout P (C fragments) -> A fragment via LDS
    __syncthreads();   // WAR vs previous iteration's reads
#pragma unroll
    for (int r = 0; r < 8; ++r) {
      plds[(r + 8 * half) * 32 + cn]      = (f16)s0[r];
      plds[(r + 8 * half) * 32 + 16 + cn] = (f16)s1[r];
    }
    __syncthreads();
    v16h pf = load_a_frag(plds, 32, lane);

    // ---- O += P @ V with the preloaded V fragments
#pragma unroll
    for (int t = 0; t < 8; ++t) o[t] = wmma16(pf, vf[t], o[t]);
  }

  // ---- normalize and store attn output (row-major [S][H*Dh], f16)
#pragma unroll
  for (int r = 0; r < 8; ++r) li[r] = 1.0f / li[r];
#pragma unroll
  for (int t = 0; t < 8; ++t)
#pragma unroll
    for (int r = 0; r < 8; ++r) {
      const float v = o[t][r] * li[r];
      const int s = q0 + r + 8 * half;
      AO[(size_t)s * DIM + h * HDIM + t * 16 + cn] = (f16)v;
    }
}

// ---------------------------------------------------------------------------
// 5) Output projection: [4096,2048] @ [2048,2048] -> f32 d_out
// ---------------------------------------------------------------------------
__global__ __launch_bounds__(128)
void gemm_out_kernel(const f16* __restrict__ A, const f16* __restrict__ Bt,
                     float* __restrict__ Out) {
  const int lane = threadIdx.x & 31;
  const int wave = threadIdx.x >> 5;
  const int n0 = blockIdx.x * 64;
  const int m0 = blockIdx.y * 64 + wave * 16;

  v8f acc[4] = {};
  gemm_strip(A + (size_t)m0 * DIM, Bt + (size_t)n0 * DIM, lane, acc);

  const int cn = lane & 15, half = lane >> 4;
#pragma unroll
  for (int t = 0; t < 4; ++t)
#pragma unroll
    for (int rr = 0; rr < 8; ++rr) {
      const int s = m0 + rr + 8 * half;
      Out[(size_t)s * DIM + n0 + t * 16 + cn] = acc[t][rr];
    }
}

// ---------------------------------------------------------------------------
extern "C" void kernel_launch(void* const* d_in, const int* in_sizes, int n_in,
                              void* d_out, int out_size, void* d_ws,
                              size_t ws_size, hipStream_t stream) {
  const float* x    = (const float*)d_in[0];   // [4096, 2048]
  const float* Wqkv = (const float*)d_in[1];   // [2048, 6144]
  const float* Wo   = (const float*)d_in[2];   // [2048, 2048]
  float* out = (float*)d_out;                  // [4096, 2048]

  char* ws = (char*)d_ws;
  size_t off = 0;
  f16* x_h   = (f16*)(ws + off); off += (size_t)SEQ * DIM * 2;        // 16 MB
  f16* WqkvT = (f16*)(ws + off); off += (size_t)QKV_N * DIM * 2;      // 24 MB
  f16* WoT   = (f16*)(ws + off); off += (size_t)DIM * DIM * 2;        //  8 MB
  f16* Qb    = (f16*)(ws + off); off += (size_t)NHEADS * SEQ * HDIM * 2;
  f16* Kb    = (f16*)(ws + off); off += (size_t)NHEADS * SEQ * HDIM * 2;
  f16* Vt    = (f16*)(ws + off); off += (size_t)NHEADS * SEQ * HDIM * 2;
  f16* AO    = (f16*)(ws + off); off += (size_t)SEQ * DIM * 2;

  const int nX = SEQ * DIM;
  cvt_f32_f16<<<(nX + 255) / 256, 256, 0, stream>>>(x, x_h, nX);

  dim3 tb(32, 8);
  transpose_f32_f16<<<dim3(QKV_N / 32, DIM / 32), tb, 0, stream>>>(
      Wqkv, WqkvT, DIM, QKV_N);
  transpose_f32_f16<<<dim3(DIM / 32, DIM / 32), tb, 0, stream>>>(
      Wo, WoT, DIM, DIM);

  gemm_qkv_kernel<<<dim3(QKV_N / 64, SEQ / 64), 128, 0, stream>>>(
      x_h, WqkvT, Qb, Kb, Vt);

  attn_kernel<<<dim3(SEQ / 16, NHEADS), 32, 0, stream>>>(Qb, Kb, Vt, AO);

  gemm_out_kernel<<<dim3(DIM / 64, SEQ / 64), 128, 0, stream>>>(AO, WoT, out);
}
